// MultiHeadSelfAttention_19464791785932
// MI455X (gfx1250) — matrix operand order repaired and checked by reading, not compiled, not run on hardware
//
#include <hip/hip_runtime.h>

// ---------------------------------------------------------------------------
// MI455X (gfx1250, wave32) multi-head self-attention, bf16 WMMA pipeline:
//   k0 : f32 -> bf16 convert (x), f32 -> bf16 transposed convert (W_qkv, W_out)
//   k1 : QKV GEMM (v_wmma_f32_16x16x32_bf16), scatters Q(scaled)/K [B,H,S,dk]
//        and V transposed [B,H,dk,S]
//   k2 : flash attention, 64-query tile / block, online softmax, WMMA for
//        Q@K^T and P@V, writes bf16 [B,S,D]
//   k3 : out-proj GEMM -> f32 d_out
// All contiguous global->LDS tile movement uses GLOBAL_LOAD_ASYNC_TO_LDS_B128
// (ASYNCcnt) when the toolchain exposes the builtin.
// ---------------------------------------------------------------------------

typedef __bf16 bf16;
typedef __bf16 v16bf __attribute__((ext_vector_type(16)));
typedef __bf16 v8bf  __attribute__((ext_vector_type(8)));
typedef float  v8f   __attribute__((ext_vector_type(8)));
typedef int    v4i   __attribute__((ext_vector_type(4)));

#define D_MODEL 1024
#define NHEADS  16
#define DK      64
#define SEQ     2048
#define BATCH   4
#define ROWS    (BATCH * SEQ)   // 8192

#if defined(__gfx1250__) && __has_builtin(__builtin_amdgcn_global_load_async_to_lds_b128)
#define HAVE_ASYNC 1
#else
#define HAVE_ASYNC 0
#endif

__device__ __forceinline__ void async_copy16(const bf16* g, bf16* l) {
#if HAVE_ASYNC
  // signature (from hipcc diagnostic): (int4 AS1* src, int4 AS3* dst, imm, imm)
  __builtin_amdgcn_global_load_async_to_lds_b128(
      (__attribute__((address_space(1))) v4i*)(v4i*)(void*)const_cast<bf16*>(g),
      (__attribute__((address_space(3))) v4i*)(v4i*)(void*)l,
      0, 0);
#else
  *(v8bf*)l = *(const v8bf*)g;
#endif
}

__device__ __forceinline__ void wait_async() {
#if HAVE_ASYNC
#if __has_builtin(__builtin_amdgcn_s_wait_asynccnt)
  __builtin_amdgcn_s_wait_asynccnt(0);
#else
  asm volatile("s_wait_asynccnt 0" ::: "memory");
#endif
#endif
}

__device__ __forceinline__ bf16 f2bf(float f) {
  unsigned u = __builtin_bit_cast(unsigned, f);
  u += 0x7fffu + ((u >> 16) & 1u);           // round-to-nearest-even
  unsigned short s = (unsigned short)(u >> 16);
  return __builtin_bit_cast(bf16, s);
}

__device__ __forceinline__ v8f vzero8() {
  v8f z;
#pragma unroll
  for (int e = 0; e < 8; ++e) z[e] = 0.0f;
  return z;
}

__device__ __forceinline__ v8f wmma_bf16(v16bf a, v16bf b, v8f c) {
  // D = A(16x32 bf16) * B(32x16 bf16) + C(16x16 f32)
  return __builtin_amdgcn_wmma_f32_16x16x32_bf16(
      /*neg_a=*/false, a, /*neg_b=*/false, b,
      /*c_mod=*/(short)0, c, /*reuse_a=*/false, /*reuse_b=*/false);
}

// A fragment (16x32 bf16): lane holds row m=lane%16; VGPR chunks:
//   k = half*8 + 0..7   and   k = 16 + half*8 + 0..7   (half = lane/16)
__device__ __forceinline__ v16bf frag_a(const bf16* p, int hal) {
  v8bf lo = *(const v8bf*)(p + hal * 8);
  v8bf hi = *(const v8bf*)(p + 16 + hal * 8);
  return __builtin_shufflevector(lo, hi, 0,1,2,3,4,5,6,7,8,9,10,11,12,13,14,15);
}
// B fragment (32x16 bf16): lane holds col n=lane%16; k = half*8 + 0..7 and 16 + half*8 + 0..7.
// p points at element [n][k0] of an LDS tile stored transposed as [N][K].
__device__ __forceinline__ v16bf frag_b(const bf16* p, int hal) {
  v8bf lo = *(const v8bf*)(p + hal * 8);
  v8bf hi = *(const v8bf*)(p + 16 + hal * 8);
  return __builtin_shufflevector(lo, hi, 0,1,2,3,4,5,6,7,8,9,10,11,12,13,14,15);
}

// --------------------------- f32 -> bf16 convert ---------------------------
__global__ __launch_bounds__(256) void cvt_f32_bf16(const float* __restrict__ in,
                                                    bf16* __restrict__ out, int n4) {
  int i = blockIdx.x * blockDim.x + threadIdx.x;
  if (i >= n4) return;
  float4 f = ((const float4*)in)[i];
  bf16* o = out + (size_t)i * 4;
  o[0] = f2bf(f.x); o[1] = f2bf(f.y); o[2] = f2bf(f.z); o[3] = f2bf(f.w);
}

// ------------------- f32 [R][C] -> bf16 [C][R] transpose -------------------
// grid = (C/64, R/64), block = 256. LDS 64x64 tile (padded rows).
#define TS 72
__global__ __launch_bounds__(256) void cvt_transpose_f32_bf16(
    const float* __restrict__ in, bf16* __restrict__ out, int R, int C) {
  __shared__ bf16 T[64 * TS];
  const int tid = threadIdx.x;
  const int c0 = blockIdx.x * 64, r0 = blockIdx.y * 64;
  // load 64x64 f32 as float4, convert, stage in LDS
#pragma unroll
  for (int i = tid; i < 1024; i += 256) {
    int r = i >> 4, c4 = (i & 15) * 4;
    float4 f = *(const float4*)&in[(size_t)(r0 + r) * C + c0 + c4];
    unsigned short s0 = __builtin_bit_cast(unsigned short, f2bf(f.x));
    unsigned short s1 = __builtin_bit_cast(unsigned short, f2bf(f.y));
    unsigned short s2 = __builtin_bit_cast(unsigned short, f2bf(f.z));
    unsigned short s3 = __builtin_bit_cast(unsigned short, f2bf(f.w));
    unsigned long long pk = (unsigned long long)s0 | ((unsigned long long)s1 << 16) |
                            ((unsigned long long)s2 << 32) | ((unsigned long long)s3 << 48);
    *(unsigned long long*)&T[r * TS + c4] = pk;
  }
  __syncthreads();
  // write transposed: contiguous 16B stores along R
#pragma unroll
  for (int i = tid; i < 512; i += 256) {
    int cc = i >> 3, rc = (i & 7) * 8;
    v8bf v;
#pragma unroll
    for (int j = 0; j < 8; ++j) v[j] = T[(rc + j) * TS + cc];
    *(v8bf*)&out[(size_t)(c0 + cc) * R + r0 + rc] = v;
  }
}

// ------------------------------- WMMA GEMM ---------------------------------
// C[M,N] = A[M,K] * Bt[N,K]^T + bias[N]   (Bt is pre-transposed weights)
// Block 256 threads (8 waves), tile BM=128 x BN=128, BK=32.
// Wave (w>>1, w&1) owns a 32x64 sub-tile: 2 (M) x 4 (N) WMMA accumulators.
// mode 0: scatter QKV -> Qo/Ko bf16 [B,H,S,DK] (Q scaled), Vo bf16 [B,H,DK,S]
// mode 1: f32 Co[M,N] += bias
#define BM 128
#define BN 128
#define BK 32

__global__ __launch_bounds__(256) void gemm_bf16(
    const bf16* __restrict__ A, const bf16* __restrict__ Bt,
    const float* __restrict__ bias, int M, int N, int K, int mode,
    bf16* __restrict__ Qo, bf16* __restrict__ Ko, bf16* __restrict__ Vo,
    float* __restrict__ Co) {
  __shared__ bf16 As[BM * BK];        // row-major [m][k]
  __shared__ bf16 Bs[BN * BK];        // [n][k]

  const int tid  = threadIdx.x;
  const int wave = tid >> 5, lane = tid & 31;
  const int hal = lane >> 4, l16 = lane & 15;
  const int bm = blockIdx.y * BM, bn = blockIdx.x * BN;
  const int waveM = (wave >> 1) * 32;
  const int waveN = (wave & 1) * 64;

  // per-thread copy assignments (2 chunks of A, 2 of B; 512 chunks each tile)
  const int rA0 = tid >> 2,           cA0 = (tid & 3) * 8;
  const int rA1 = (tid + 256) >> 2,   cA1 = ((tid + 256) & 3) * 8;
  const bf16* pA0 = &A[(size_t)(bm + rA0) * K + cA0];
  const bf16* pA1 = &A[(size_t)(bm + rA1) * K + cA1];
  const bf16* pB0 = &Bt[(size_t)(bn + rA0) * K + cA0];
  const bf16* pB1 = &Bt[(size_t)(bn + rA1) * K + cA1];
  bf16* lA0 = &As[rA0 * BK + cA0];
  bf16* lA1 = &As[rA1 * BK + cA1];
  bf16* lB0 = &Bs[rA0 * BK + cA0];
  bf16* lB1 = &Bs[rA1 * BK + cA1];

  v8f acc[2][4];
#pragma unroll
  for (int mt = 0; mt < 2; ++mt)
#pragma unroll
    for (int nt = 0; nt < 4; ++nt) acc[mt][nt] = vzero8();

  for (int k0 = 0; k0 < K; k0 += BK) {
    if (k0 + BK < K) {
      __builtin_prefetch(pA0 + BK, 0, 3);
      __builtin_prefetch(pB0 + BK, 0, 3);
    }
    async_copy16(pA0, lA0);
    async_copy16(pA1, lA1);
    async_copy16(pB0, lB0);
    async_copy16(pB1, lB1);
    pA0 += BK; pA1 += BK; pB0 += BK; pB1 += BK;
    wait_async();
    __syncthreads();

    v16bf af[2], bfr[4];
#pragma unroll
    for (int mt = 0; mt < 2; ++mt)
      af[mt] = frag_a(&As[(waveM + mt * 16 + l16) * BK], hal);
#pragma unroll
    for (int nt = 0; nt < 4; ++nt)
      bfr[nt] = frag_b(&Bs[(waveN + nt * 16 + l16) * BK], hal);
#pragma unroll
    for (int mt = 0; mt < 2; ++mt)
#pragma unroll
      for (int nt = 0; nt < 4; ++nt)
        acc[mt][nt] = wmma_bf16(af[mt], bfr[nt], acc[mt][nt]);
    __syncthreads();
  }

  // Epilogue. C layout: lane l16 = col-within-16, VGPR r -> row r + 8*hal.
#pragma unroll
  for (int mt = 0; mt < 2; ++mt) {
#pragma unroll
    for (int nt = 0; nt < 4; ++nt) {
#pragma unroll
      for (int r = 0; r < 8; ++r) {
        int row = bm + waveM + mt * 16 + r + hal * 8;
        int col = bn + waveN + nt * 16 + l16;
        float v = acc[mt][nt][r] + bias[col];
        if (mode == 0) {
          int which = col >> 10;          // 0=Q 1=K 2=V
          int h = (col >> 6) & (NHEADS - 1);
          int d = col & (DK - 1);
          int b = row >> 11;              // row / SEQ
          int s = row & (SEQ - 1);
          size_t hb = (size_t)b * NHEADS + h;
          if (which == 0)      Qo[(hb * SEQ + s) * DK + d] = f2bf(v * 0.125f);
          else if (which == 1) Ko[(hb * SEQ + s) * DK + d] = f2bf(v);
          else                 Vo[(hb * DK + d) * SEQ + s] = f2bf(v); // transposed
        } else {
          Co[(size_t)row * N + col] = v;
        }
      }
    }
  }
}

// ---------------------------- flash attention ------------------------------
// grid = (SEQ/64, BATCH*NHEADS), block = 128 (4 waves).
// Each block: 64 query rows; each wave: 16 query rows. Stream 64-key tiles.
// Q,K are [B,H,S,DK]; V is [B,H,DK,S] (pre-transposed) -> all tiles contiguous.
__global__ __launch_bounds__(128) void flash_attn(
    const bf16* __restrict__ Q, const bf16* __restrict__ Kh_,
    const bf16* __restrict__ Vt, bf16* __restrict__ O) {
  __shared__ bf16 Qs[64 * 64];     // [m][d]  (A-friendly)
  __shared__ bf16 Ks[64 * 64];     // [t][d]  (B-friendly for Q@K^T: n=t,k=d)
  __shared__ bf16 Vs[64 * 64];     // [d][t]  (B-friendly for P@V:  n=d,k=t)
  __shared__ bf16 Ps[4 * 16 * 64]; // per-wave P tile [m][t]

  const int tid = threadIdx.x;
  const int wave = tid >> 5, lane = tid & 31;
  const int hal = lane >> 4, l16 = lane & 15;
  const size_t head = blockIdx.y;            // b*NHEADS + h
  const int q0 = blockIdx.x * 64;

  const bf16* Qh = Q   + head * SEQ * DK;
  const bf16* Kp = Kh_ + head * SEQ * DK;
  const bf16* Vp = Vt  + head * DK * SEQ;    // [d][s]

  // load Q tile (contiguous 64x64 bf16)
#pragma unroll
  for (int i = tid; i < 512; i += 128)
    async_copy16(&Qh[(size_t)q0 * DK + i * 8], &Qs[i * 8]);

  v8f o[4];
  float mi[8], li[8];
#pragma unroll
  for (int nt = 0; nt < 4; ++nt) o[nt] = vzero8();
#pragma unroll
  for (int r = 0; r < 8; ++r) { mi[r] = -1e30f; li[r] = 0.0f; }

  for (int t0 = 0; t0 < SEQ; t0 += 64) {
    // K tile rows [t][d] (contiguous)
#pragma unroll
    for (int i = tid; i < 512; i += 128)
      async_copy16(&Kp[(size_t)t0 * DK + i * 8], &Ks[i * 8]);
    // V tile rows [d][t] (contiguous thanks to pre-transposed V)
#pragma unroll
    for (int i = tid; i < 512; i += 128) {
      int d = i >> 3, tc = (i & 7) * 8;
      async_copy16(&Vp[(size_t)d * SEQ + t0 + tc], &Vs[d * 64 + tc]);
    }
    wait_async();
    __syncthreads();

    // S = Q @ K^T : 16 rows x 64 keys per wave, K-dim = DK = 2 wmma steps
    v8f sc[4];
#pragma unroll
    for (int nt = 0; nt < 4; ++nt) sc[nt] = vzero8();
#pragma unroll
    for (int kk = 0; kk < 2; ++kk) {
      int k0 = kk * 32;
      v16bf a = frag_a(&Qs[(wave * 16 + l16) * 64 + k0], hal);
#pragma unroll
      for (int nt = 0; nt < 4; ++nt) {
        v16bf b = frag_b(&Ks[(nt * 16 + l16) * 64 + k0], hal);
        sc[nt] = wmma_bf16(a, b, sc[nt]);
      }
    }

    // online softmax per row (row r+8*hal, cols spread over 16 lanes x 4 tiles)
#pragma unroll
    for (int r = 0; r < 8; ++r) {
      float mx = fmaxf(fmaxf(sc[0][r], sc[1][r]), fmaxf(sc[2][r], sc[3][r]));
#pragma unroll
      for (int off = 1; off <= 8; off <<= 1) mx = fmaxf(mx, __shfl_xor(mx, off, 32));
      float mnew = fmaxf(mi[r], mx);
      float p0 = __expf(sc[0][r] - mnew);
      float p1 = __expf(sc[1][r] - mnew);
      float p2 = __expf(sc[2][r] - mnew);
      float p3 = __expf(sc[3][r] - mnew);
      float ps = p0 + p1 + p2 + p3;
#pragma unroll
      for (int off = 1; off <= 8; off <<= 1) ps += __shfl_xor(ps, off, 32);
      float alpha = __expf(mi[r] - mnew);
      li[r] = li[r] * alpha + ps;
      mi[r] = mnew;
#pragma unroll
      for (int nt = 0; nt < 4; ++nt) o[nt][r] *= alpha;
      int m = wave * 16 + r + hal * 8;
      Ps[m * 64 +  0 + l16] = f2bf(p0);
      Ps[m * 64 + 16 + l16] = f2bf(p1);
      Ps[m * 64 + 32 + l16] = f2bf(p2);
      Ps[m * 64 + 48 + l16] = f2bf(p3);
    }
    // O += P @ V  (same-wave LDS RAW: DS ops are in-order per wave)
#pragma unroll
    for (int kk = 0; kk < 2; ++kk) {
      int k0 = kk * 32;
      v16bf a = frag_a(&Ps[(wave * 16 + l16) * 64 + k0], hal);
#pragma unroll
      for (int nt = 0; nt < 4; ++nt) {
        v16bf b = frag_b(&Vs[(nt * 16 + l16) * 64 + k0], hal);
        o[nt] = wmma_bf16(a, b, o[nt]);
      }
    }
    __syncthreads();
  }

  // write O (bf16) into [B,S,D] = [b*SEQ+s][h*DK+d]
  const size_t b = head >> 4, h = head & (NHEADS - 1);
#pragma unroll
  for (int nt = 0; nt < 4; ++nt) {
#pragma unroll
    for (int r = 0; r < 8; ++r) {
      int m = wave * 16 + r + hal * 8;
      int d = nt * 16 + l16;
      float v = o[nt][r] / li[r];
      O[((b * SEQ + (q0 + m)) * D_MODEL) + h * DK + d] = f2bf(v);
    }
  }
}

// ------------------------------- launcher ----------------------------------
extern "C" void kernel_launch(void* const* d_in, const int* in_sizes, int n_in,
                              void* d_out, int out_size, void* d_ws, size_t ws_size,
                              hipStream_t stream) {
  const float* x     = (const float*)d_in[0];
  const float* W_qkv = (const float*)d_in[1];
  const float* b_qkv = (const float*)d_in[2];
  const float* W_out = (const float*)d_in[3];
  const float* b_out = (const float*)d_in[4];
  float* out = (float*)d_out;

  char* ws = (char*)d_ws;
  size_t off = 0;
  auto carve = [&](size_t bytes) {
    void* p = ws + off;
    off += (bytes + 255) & ~(size_t)255;
    return p;
  };
  const size_t nX    = (size_t)ROWS * D_MODEL;        // 8.39M
  const size_t nWq   = (size_t)D_MODEL * 3 * D_MODEL; // 3.15M
  const size_t nWo   = (size_t)D_MODEL * D_MODEL;     // 1.05M
  const size_t nHead = (size_t)BATCH * NHEADS * SEQ * DK;

  bf16* xb    = (bf16*)carve(nX * 2);
  bf16* wqkvt = (bf16*)carve(nWq * 2);   // [3*D_MODEL][D_MODEL]
  bf16* woutt = (bf16*)carve(nWo * 2);   // [D_MODEL][D_MODEL]
  bf16* Qb    = (bf16*)carve(nHead * 2); // [B,H,S,DK]
  bf16* Kb    = (bf16*)carve(nHead * 2); // [B,H,S,DK]
  bf16* Vb    = (bf16*)carve(nHead * 2); // [B,H,DK,S]
  bf16* attnb = (bf16*)carve(nX * 2);
  (void)ws_size; // ~92 MB used

  // converts (x: straight; weights: transposed to [N][K])
  cvt_f32_bf16<<<dim3((unsigned)(nX / 4 + 255) / 256), 256, 0, stream>>>(x, xb, (int)(nX / 4));
  cvt_transpose_f32_bf16<<<dim3(3 * D_MODEL / 64, D_MODEL / 64), 256, 0, stream>>>(
      W_qkv, wqkvt, D_MODEL, 3 * D_MODEL);
  cvt_transpose_f32_bf16<<<dim3(D_MODEL / 64, D_MODEL / 64), 256, 0, stream>>>(
      W_out, woutt, D_MODEL, D_MODEL);

  // QKV projection + scatter (Q pre-scaled by 1/sqrt(dk), V transposed)
  gemm_bf16<<<dim3(3 * D_MODEL / BN, ROWS / BM), 256, 0, stream>>>(
      xb, wqkvt, b_qkv, ROWS, 3 * D_MODEL, D_MODEL, /*mode=*/0,
      Qb, Kb, Vb, nullptr);

  // flash attention
  flash_attn<<<dim3(SEQ / 64, BATCH * NHEADS), 128, 0, stream>>>(Qb, Kb, Vb, attnb);

  // output projection -> f32
  gemm_bf16<<<dim3(D_MODEL / BN, ROWS / BM), 256, 0, stream>>>(
      attnb, woutt, b_out, ROWS, D_MODEL, D_MODEL, /*mode=*/1,
      nullptr, nullptr, nullptr, out);
}
